// MVATModel_36756330119359
// MI455X (gfx1250) — matrix operand order repaired and checked by reading, not compiled, not run on hardware
//
#include <hip/hip_runtime.h>
#include <hip/hip_bf16.h>

#define BATCH   8
#define SEQ     1024
#define DIM     768
#define DIM3    2304
#define DIM2X   1536
#define NHEADS  8
#define HDIM    96
#define LAYERS  3
#define KNN     5
#define ROWS    (BATCH * SEQ)          // 8192
#define KTILES  (SEQ / 16)             // 64

typedef __attribute__((ext_vector_type(16))) _Float16     v16h;
typedef __attribute__((ext_vector_type(8)))  _Float16     v8h;
typedef __attribute__((ext_vector_type(8)))  float        v8f;
typedef __attribute__((ext_vector_type(4)))  unsigned int v4u;
typedef __attribute__((ext_vector_type(8)))  int          v8i;
typedef __attribute__((ext_vector_type(4)))  int          v4i;

#if defined(__has_builtin)
#if __has_builtin(__builtin_amdgcn_tensor_load_to_lds) && \
    __has_builtin(__builtin_amdgcn_s_wait_tensorcnt)
#define HAVE_TDM 1
#endif
#endif
#ifndef HAVE_TDM
#define HAVE_TDM 0
#endif

// ---------------------------------------------------------------------------
// WMMA fragment helpers (wave32, contiguous k order k = 16*(lane/16) + s, used by A and B alike)
// A 16x32 f16: lane<16 -> M=lane, K={0..15}; lane>=16 -> M=lane-16,
// K={16..31}.
// ---------------------------------------------------------------------------
__device__ inline v16h load_frag_a(const _Float16* __restrict__ A, int lda) {
    int l = threadIdx.x & 31;
    int m = l & 15;
    int koff = (l >> 4) << 4;
    const _Float16* row = A + (size_t)m * lda + koff;
    v8h lo = *reinterpret_cast<const v8h*>(row);
    v8h hi = *reinterpret_cast<const v8h*>(row + 8);
    v16h f;
#pragma unroll
    for (int e = 0; e < 8; ++e) { f[e] = lo[e]; f[e + 8] = hi[e]; }
    return f;
}

// B fragment where B[k][n] = X[n][k] (X row-major [N][K]): contiguous per lane.
__device__ inline v16h load_frag_bt(const _Float16* __restrict__ X, int ld) {
    int l = threadIdx.x & 31;
    int n = l & 15;
    int kb = (l >> 4) << 4;
    return *reinterpret_cast<const v16h*>(X + (size_t)n * ld + kb);
}

// ---------------------------------------------------------------------------
// conversions: f32->f16, and f32 [K,N] -> f16 transposed [N,K]
// ---------------------------------------------------------------------------
__global__ void k_f32_to_f16(const float* __restrict__ src, _Float16* __restrict__ dst, int n) {
    int i = blockIdx.x * blockDim.x + threadIdx.x;
    if (i < n) dst[i] = (_Float16)src[i];
}

__global__ void k_w_to_bt16(const float* __restrict__ w, _Float16* __restrict__ wt,
                            int K, int N) {
    int i = blockIdx.x * blockDim.x + threadIdx.x;
    if (i >= K * N) return;
    int n = i / K, k = i - n * K;
    wt[(size_t)n * K + k] = (_Float16)w[(size_t)k * N + n];
}

// ---------------------------------------------------------------------------
// WMMA GEMM: C[M,N] = A[M,K] @ BT[N,K]^T + bias.  One wave -> 32x64 tile.
// ---------------------------------------------------------------------------
template <typename OutT>
__global__ void k_gemm_wmma(const _Float16* __restrict__ A, int lda,
                            const _Float16* __restrict__ BT, int ldb,   // ldb == K
                            const float* __restrict__ bias,
                            OutT* __restrict__ C, int ldc,
                            int M, int N, int K) {
    int wave = (int)((blockIdx.x * (size_t)blockDim.x + threadIdx.x) >> 5);
    int mTiles = M >> 5;
    int nTiles = N >> 6;
    if (wave >= mTiles * nTiles) return;
    int mT = wave % mTiles;
    int nT = wave / mTiles;
    int l = threadIdx.x & 31;
    const _Float16* Ab = A + (size_t)(mT << 5) * lda;
    int n0 = nT << 6;

    v8f acc[2][4] = {};
    for (int k0 = 0; k0 < K; k0 += 32) {
        v16h a0 = load_frag_a(Ab + k0, lda);
        v16h a1 = load_frag_a(Ab + (size_t)16 * lda + k0, lda);
        if (k0 + 32 < K)
            __builtin_prefetch(Ab + (size_t)(l & 15) * lda + k0 + 32, 0, 1);
#pragma unroll
        for (int j = 0; j < 4; ++j) {
            v16h b = load_frag_bt(BT + (size_t)(n0 + j * 16) * ldb + k0, ldb);
            acc[0][j] = __builtin_amdgcn_wmma_f32_16x16x32_f16(
                false, a0, false, b, (short)0, acc[0][j], false, false);
            acc[1][j] = __builtin_amdgcn_wmma_f32_16x16x32_f16(
                false, a1, false, b, (short)0, acc[1][j], false, false);
        }
    }
    int n = l & 15;
    int mb = (l >> 4) << 3;
#pragma unroll
    for (int j = 0; j < 4; ++j) {
        float bv = bias ? bias[n0 + j * 16 + n] : 0.f;
#pragma unroll
        for (int half = 0; half < 2; ++half)
#pragma unroll
            for (int r = 0; r < 8; ++r) {
                int row = (mT << 5) + half * 16 + mb + r;
                C[(size_t)row * ldc + n0 + j * 16 + n] = (OutT)(acc[half][j][r] + bv);
            }
    }
}

// ---------------------------------------------------------------------------
// Similarity + per-row top-5.  One wave per (b, 16-row tile).
// ---------------------------------------------------------------------------
__global__ void k_sim_topk(const _Float16* __restrict__ x16, int* __restrict__ idx) {
    int blk = blockIdx.x;
    int qt = blk & (KTILES - 1);
    int b  = blk / KTILES;
    const _Float16* X = x16 + (size_t)b * SEQ * DIM;
    int l = threadIdx.x & 31;

    float topv[KNN];
    int   topi[KNN];
#pragma unroll
    for (int p = 0; p < KNN; ++p) { topv[p] = -3.0e38f; topi[p] = 0; }

    __shared__ float sLds[16 * 16];

    for (int kt = 0; kt < KTILES; ++kt) {
        v8f sc = {};
#pragma unroll 4
        for (int c = 0; c < DIM / 32; ++c) {
            v16h a  = load_frag_a (X + (size_t)(qt * 16) * DIM + c * 32, DIM);
            v16h bt = load_frag_bt(X + (size_t)(kt * 16) * DIM + c * 32, DIM);
            sc = __builtin_amdgcn_wmma_f32_16x16x32_f16(
                false, a, false, bt, (short)0, sc, false, false);
        }
        int n = l & 15, mb = (l >> 4) << 3;
#pragma unroll
        for (int r = 0; r < 8; ++r) sLds[(mb + r) * 16 + n] = sc[r];
        __syncthreads();
        if (l < 16) {
            for (int j = 0; j < 16; ++j) {
                float v = sLds[l * 16 + j];
                int   t = kt * 16 + j;
#pragma unroll
                for (int p = 0; p < KNN; ++p) {
                    if (v > topv[p]) {
                        float tv = topv[p]; int ti = topi[p];
                        topv[p] = v; topi[p] = t; v = tv; t = ti;
                    }
                }
            }
        }
        __syncthreads();
    }
    if (l < 16) {
        size_t base = ((size_t)b * SEQ + qt * 16 + l) * KNN;
#pragma unroll
        for (int p = 0; p < KNN; ++p) idx[base + p] = topi[p];
    }
}

// ---------------------------------------------------------------------------
// Adjacency bitmask (1024 bits/row), symmetrize + self loops + degrees
// ---------------------------------------------------------------------------
__global__ void k_build_adj(const int* __restrict__ idx, unsigned int* __restrict__ adj) {
    int t = blockIdx.x * blockDim.x + threadIdx.x;
    if (t >= BATCH * SEQ * KNN) return;
    int s = (t / KNN) & (SEQ - 1);
    int b = t / (KNN * SEQ);
    int nb = idx[t];
    unsigned int* A = adj + (size_t)b * SEQ * (SEQ / 32);
    atomicOr(&A[(size_t)s * 32 + (nb >> 5)], 1u << (nb & 31));
    atomicOr(&A[(size_t)nb * 32 + (s >> 5)], 1u << (s & 31));
}

__global__ void k_add_self(unsigned int* __restrict__ adj) {
    int t = blockIdx.x * blockDim.x + threadIdx.x;
    if (t >= ROWS) return;
    int s = t & (SEQ - 1);
    adj[(size_t)t * 32 + (s >> 5)] |= (1u << (s & 31));
}

__global__ void k_degrees(const unsigned int* __restrict__ adj, float* __restrict__ dinv) {
    int t = blockIdx.x * blockDim.x + threadIdx.x;
    if (t >= ROWS) return;
    const unsigned int* row = adj + (size_t)t * 32;
    int deg = 0;
#pragma unroll
    for (int w = 0; w < 32; ++w) deg += __popc(row[w]);
    dinv[t] = rsqrtf((float)deg);
}

// ---------------------------------------------------------------------------
// Sparse GCN aggregation: block per (b,s)
// ---------------------------------------------------------------------------
__global__ void k_gcn_agg(const unsigned int* __restrict__ adj,
                          const float* __restrict__ dinv,
                          const float* __restrict__ tmp,
                          const float* __restrict__ bias,
                          float* __restrict__ outp) {
    int bs = blockIdx.x;
    int b = bs >> 10;
    const unsigned int* row = adj + (size_t)bs * 32;
    float ds = dinv[bs];
    int d0 = threadIdx.x;
    float a0 = 0.f, a1 = 0.f, a2 = 0.f;
    for (int w = 0; w < 32; ++w) {
        unsigned int bits = row[w];
        while (bits) {
            int bit = __ffs(bits) - 1;
            bits &= bits - 1;
            int t = w * 32 + bit;
            float c = ds * dinv[(b << 10) + t];
            const float* tr = tmp + ((size_t)(b << 10) + t) * DIM;
            a0 += c * tr[d0];
            a1 += c * tr[d0 + 256];
            a2 += c * tr[d0 + 512];
        }
    }
    float* orow = outp + (size_t)bs * DIM;
    orow[d0]       = a0 + bias[d0];
    orow[d0 + 256] = a1 + bias[d0 + 256];
    orow[d0 + 512] = a2 + bias[d0 + 512];
}

// ---------------------------------------------------------------------------
// Flash attention: one wave per (b, head, 16-query tile).
// Scores: 3 WMMAs (K^T frags are contiguous loads). PV: TDM-staged V tile in
// LDS (tensor_load_to_lds + s_wait_tensorcnt), then 6 WMMAs per key tile.
// ---------------------------------------------------------------------------
__global__ void k_flash_attn(const _Float16* __restrict__ qkv16, _Float16* __restrict__ ao16) {
    int blk = blockIdx.x;
    int qt = blk & (KTILES - 1);
    int h  = (blk >> 6) & (NHEADS - 1);
    int b  = blk >> 9;
    const int ld = DIM3;
    const _Float16* base = qkv16 + (size_t)b * SEQ * ld;
    const _Float16* Q = base + h * HDIM;
    const _Float16* K = base + DIM + h * HDIM;
    const _Float16* V = base + 2 * DIM + h * HDIM;
    int l = threadIdx.x & 31;
    const float scale = 0.1020620726f;   // 1/sqrt(96)

    v16h qf[3];
#pragma unroll
    for (int c = 0; c < 3; ++c)
        qf[c] = load_frag_a(Q + (size_t)(qt * 16) * ld + c * 32, ld);

    v8f o[6] = {};
    float mrun[8], lrun[8];
#pragma unroll
    for (int r = 0; r < 8; ++r) { mrun[r] = -3.0e38f; lrun[r] = 0.f; }

    __shared__ _Float16 pLds[16 * 16];
#if HAVE_TDM
    __shared__ _Float16 sV[16 * HDIM];           // 16 x 96 staged V tile
    unsigned ldsV = (unsigned)(size_t)(void*)&sV[0];
#endif

    for (int kt = 0; kt < KTILES; ++kt) {
#if HAVE_TDM
        {   // TDM: DMA V rows [kt*16, kt*16+16) x 96 cols into LDS (ISA D#, 2D)
            unsigned long long ga =
                (unsigned long long)(size_t)(V + (size_t)(kt * 16) * ld);
            v4u g0 = { 1u,                                    // count=1
                       ldsV,                                  // lds_addr
                       (unsigned)ga,                          // global_addr lo
                       ((unsigned)(ga >> 32) & 0x01FFFFFFu) | 0x80000000u }; // hi | type=2
            v8i g1 = { 0x00010000,            // data_size = 2 bytes
                       (int)(HDIM << 16),     // tensor_dim0 = 96
                       (int)(16   << 16),     // tensor_dim1 = 16
                       (int)(HDIM << 16),     // tile_dim0  = 96
                       16,                    // tile_dim1  = 16
                       DIM3,                  // tensor_dim0_stride = 2304
                       0, 0 };
            v4i g2 = { 0, 0, 0, 0 };
            v8i gz = { 0, 0, 0, 0, 0, 0, 0, 0 };
            __builtin_amdgcn_tensor_load_to_lds(g0, g1, g2, g2, gz, 0);
        }
#endif
        v8f sc = {};
#pragma unroll
        for (int c = 0; c < 3; ++c) {
            v16h bf = load_frag_bt(K + (size_t)(kt * 16) * ld + c * 32, ld);
            sc = __builtin_amdgcn_wmma_f32_16x16x32_f16(
                false, qf[c], false, bf, (short)0, sc, false, false);
        }
        float rmax[8], rsum[8];
#pragma unroll
        for (int r = 0; r < 8; ++r) { sc[r] *= scale; rmax[r] = sc[r]; }
#pragma unroll
        for (int off = 1; off < 16; off <<= 1)
#pragma unroll
            for (int r = 0; r < 8; ++r)
                rmax[r] = fmaxf(rmax[r], __shfl_xor(rmax[r], off, 32));
#pragma unroll
        for (int r = 0; r < 8; ++r) {
            float mnew = fmaxf(mrun[r], rmax[r]);
            float sf = __expf(mrun[r] - mnew);
            mrun[r] = mnew;
            float p = __expf(sc[r] - mnew);
            sc[r] = p;
            rsum[r] = p;
            lrun[r] *= sf;
#pragma unroll
            for (int nt = 0; nt < 6; ++nt) o[nt][r] *= sf;
        }
#pragma unroll
        for (int off = 1; off < 16; off <<= 1)
#pragma unroll
            for (int r = 0; r < 8; ++r)
                rsum[r] += __shfl_xor(rsum[r], off, 32);
#pragma unroll
        for (int r = 0; r < 8; ++r) lrun[r] += rsum[r];

        // stash P tile, re-read in A-fragment layout (zero-padded to K=32)
        int n = l & 15, mb = (l >> 4) << 3;
#pragma unroll
        for (int r = 0; r < 8; ++r) pLds[(mb + r) * 16 + n] = (_Float16)sc[r];
        __syncthreads();
        v16h pa;
        {
            int m = l & 15, koff = (l >> 4) << 4;
#pragma unroll
            for (int e = 0; e < 16; ++e) {
                int kk = koff + e;
                pa[e] = (kk < 16) ? pLds[m * 16 + kk] : (_Float16)0.f;
            }
        }
        __syncthreads();
#if HAVE_TDM
        __builtin_amdgcn_s_wait_tensorcnt(0);
#endif
#pragma unroll
        for (int nt = 0; nt < 6; ++nt) {
            v16h vf;
            int nn = l & 15, kb = (l >> 4) << 4;
#pragma unroll
            for (int e = 0; e < 16; ++e) {
                int kk = kb + e;
#if HAVE_TDM
                vf[e] = (kk < 16) ? sV[kk * HDIM + nt * 16 + nn] : (_Float16)0.f;
#else
                vf[e] = (kk < 16) ? V[(size_t)(kt * 16 + kk) * ld + nt * 16 + nn]
                                  : (_Float16)0.f;
#endif
            }
            o[nt] = __builtin_amdgcn_wmma_f32_16x16x32_f16(
                false, pa, false, vf, (short)0, o[nt], false, false);
        }
    }
    int n = l & 15, mb = (l >> 4) << 3;
#pragma unroll
    for (int nt = 0; nt < 6; ++nt)
#pragma unroll
        for (int r = 0; r < 8; ++r) {
            size_t row = (size_t)b * SEQ + qt * 16 + mb + r;
            ao16[row * DIM + h * HDIM + nt * 16 + n] = (_Float16)(o[nt][r] / lrun[r]);
        }
}

// ---------------------------------------------------------------------------
// Build f16 concat [gcn | attn]  ([M, 1536])
// ---------------------------------------------------------------------------
__global__ void k_make_cat16(const float* __restrict__ g, const float* __restrict__ a,
                             _Float16* __restrict__ cat) {
    int i = blockIdx.x * blockDim.x + threadIdx.x;
    if (i >= ROWS * DIM) return;
    int row = i / DIM, d = i - row * DIM;
    cat[(size_t)row * DIM2X + d]       = (_Float16)g[i];
    cat[(size_t)row * DIM2X + DIM + d] = (_Float16)a[i];
}

// ---------------------------------------------------------------------------
// sigmoid gate + fuse + residual + LayerNorm (block per row)
// ---------------------------------------------------------------------------
__global__ void k_fuse_ln(const float* __restrict__ gate_lin,
                          const float* __restrict__ gcn,
                          const float* __restrict__ attn,
                          const float* __restrict__ ln_s,
                          const float* __restrict__ ln_b,
                          float* __restrict__ h) {
    __shared__ float red[256];
    int row = blockIdx.x, t = threadIdx.x;
    size_t base = (size_t)row * DIM;
    float y[3];
#pragma unroll
    for (int i = 0; i < 3; ++i) {
        int d = t + i * 256;
        float g = 1.f / (1.f + __expf(-gate_lin[base + d]));
        y[i] = g * gcn[base + d] + (1.f - g) * attn[base + d] + h[base + d];
    }
    float s = y[0] + y[1] + y[2];
    red[t] = s; __syncthreads();
    for (int o = 128; o > 0; o >>= 1) { if (t < o) red[t] += red[t + o]; __syncthreads(); }
    float mu = red[0] * (1.f / DIM);
    __syncthreads();
    float v = 0.f;
#pragma unroll
    for (int i = 0; i < 3; ++i) { float d0 = y[i] - mu; v += d0 * d0; }
    red[t] = v; __syncthreads();
    for (int o = 128; o > 0; o >>= 1) { if (t < o) red[t] += red[t + o]; __syncthreads(); }
    float inv = rsqrtf(red[0] * (1.f / DIM) + 1e-5f);
#pragma unroll
    for (int i = 0; i < 3; ++i) {
        int d = t + i * 256;
        h[base + d] = (y[i] - mu) * inv * ln_s[d] + ln_b[d];
    }
}

// ---------------------------------------------------------------------------
// Host orchestration
// ---------------------------------------------------------------------------
template <typename OutT>
static void launch_gemm(const _Float16* A, int lda, const _Float16* BT, int ldb,
                        const float* bias, OutT* C, int ldc,
                        int M, int N, int K, hipStream_t stream) {
    int waves = (M >> 5) * (N >> 6);
    int blocks = (waves + 7) / 8;
    k_gemm_wmma<OutT><<<blocks, 256, 0, stream>>>(A, lda, BT, ldb, bias, C, ldc, M, N, K);
}

extern "C" void kernel_launch(void* const* d_in, const int* in_sizes, int n_in,
                              void* d_out, int out_size, void* d_ws, size_t ws_size,
                              hipStream_t stream) {
    const float* x          = (const float*)d_in[0];
    const float* gcn_w      = (const float*)d_in[1];
    const float* gcn_b      = (const float*)d_in[2];
    const float* attn_in_w  = (const float*)d_in[3];
    const float* attn_in_b  = (const float*)d_in[4];
    const float* attn_out_w = (const float*)d_in[5];
    const float* attn_out_b = (const float*)d_in[6];
    const float* gate_w     = (const float*)d_in[7];
    const float* gate_b     = (const float*)d_in[8];
    const float* ln_scale   = (const float*)d_in[9];
    const float* ln_bias    = (const float*)d_in[10];
    const float* proj_w     = (const float*)d_in[11];
    const float* proj_b     = (const float*)d_in[12];

    char* ws = (char*)d_ws;
    size_t off = 0;
    auto alloc = [&](size_t bytes) -> void* {
        void* p = ws + off;
        off += (bytes + 255) & ~(size_t)255;
        return p;
    };
    float*        h       = (float*)       alloc((size_t)ROWS * DIM * 4);
    _Float16*     h16     = (_Float16*)    alloc((size_t)ROWS * DIM * 2);
    _Float16*     wt16    = (_Float16*)    alloc((size_t)DIM * DIM3 * 2);   // transposed weights
    float*        tmp     = (float*)       alloc((size_t)ROWS * DIM * 4);   // also gate_lin
    float*        gcn_out = (float*)       alloc((size_t)ROWS * DIM * 4);
    _Float16*     qkv16   = (_Float16*)    alloc((size_t)ROWS * DIM3 * 2);
    _Float16*     ao16    = (_Float16*)    alloc((size_t)ROWS * DIM * 2);
    float*        attn_o  = (float*)       alloc((size_t)ROWS * DIM * 4);
    _Float16*     cat16   = (_Float16*)    alloc((size_t)ROWS * DIM2X * 2);
    unsigned int* adj     = (unsigned int*)alloc((size_t)ROWS * 32 * 4);
    float*        dinv    = (float*)       alloc((size_t)ROWS * 4);
    int*          idx     = (int*)         alloc((size_t)ROWS * KNN * 4);
    (void)ws_size; (void)in_sizes; (void)n_in; (void)out_size;

    const int NE = ROWS * DIM;
    const int CB = 256;

    // h = x (f32, keep input immutable), h16 = f16(x)
    (void)hipMemcpyAsync(h, x, (size_t)NE * 4, hipMemcpyDeviceToDevice, stream);
    k_f32_to_f16<<<(NE + CB - 1) / CB, CB, 0, stream>>>(x, h16, NE);

    // graph build (once, from input embeddings)
    (void)hipMemsetAsync(adj, 0, (size_t)ROWS * 32 * 4, stream);
    k_sim_topk<<<BATCH * KTILES, 32, 0, stream>>>(h16, idx);
    k_build_adj<<<(ROWS * KNN + CB - 1) / CB, CB, 0, stream>>>(idx, adj);
    k_add_self<<<(ROWS + CB - 1) / CB, CB, 0, stream>>>(adj);
    k_degrees<<<(ROWS + CB - 1) / CB, CB, 0, stream>>>(adj, dinv);

    for (int l = 0; l < LAYERS; ++l) {
        // --- GCN branch: tmp = h @ gcn_w[l]; gcn_out = A_norm @ tmp + b
        k_w_to_bt16<<<(DIM * DIM + CB - 1) / CB, CB, 0, stream>>>(
            gcn_w + (size_t)l * DIM * DIM, wt16, DIM, DIM);
        launch_gemm<float>(h16, DIM, wt16, DIM, nullptr, tmp, DIM, ROWS, DIM, DIM, stream);
        k_gcn_agg<<<ROWS, 256, 0, stream>>>(adj, dinv, tmp, gcn_b + (size_t)l * DIM, gcn_out);

        // --- Attention: qkv16 = f16(h @ w_in + b_in); flash; out-proj
        k_w_to_bt16<<<(DIM * DIM3 + CB - 1) / CB, CB, 0, stream>>>(
            attn_in_w + (size_t)l * DIM * DIM3, wt16, DIM, DIM3);
        launch_gemm<_Float16>(h16, DIM, wt16, DIM, attn_in_b + (size_t)l * DIM3,
                              qkv16, DIM3, ROWS, DIM3, DIM, stream);
        k_flash_attn<<<BATCH * NHEADS * KTILES, 32, 0, stream>>>(qkv16, ao16);
        k_w_to_bt16<<<(DIM * DIM + CB - 1) / CB, CB, 0, stream>>>(
            attn_out_w + (size_t)l * DIM * DIM, wt16, DIM, DIM);
        launch_gemm<float>(ao16, DIM, wt16, DIM, attn_out_b + (size_t)l * DIM,
                           attn_o, DIM, ROWS, DIM, DIM, stream);

        // --- Gate: gate_lin = [gcn|attn] @ gate_w[l] + gate_b
        k_make_cat16<<<(NE + CB - 1) / CB, CB, 0, stream>>>(gcn_out, attn_o, cat16);
        k_w_to_bt16<<<(DIM2X * DIM + CB - 1) / CB, CB, 0, stream>>>(
            gate_w + (size_t)l * DIM2X * DIM, wt16, DIM2X, DIM);
        launch_gemm<float>(cat16, DIM2X, wt16, DIM2X, gate_b + (size_t)l * DIM,
                           tmp, DIM, ROWS, DIM, DIM2X, stream);

        // --- fuse + residual + LN (in place on h), refresh h16
        k_fuse_ln<<<ROWS, 256, 0, stream>>>(tmp, gcn_out, attn_o,
                                            ln_scale + (size_t)l * DIM,
                                            ln_bias + (size_t)l * DIM, h);
        k_f32_to_f16<<<(NE + CB - 1) / CB, CB, 0, stream>>>(h, h16, NE);
    }

    // final projection -> d_out (f32)
    k_w_to_bt16<<<(DIM * DIM + CB - 1) / CB, CB, 0, stream>>>(proj_w, wt16, DIM, DIM);
    launch_gemm<float>(h16, DIM, wt16, DIM, proj_b, (float*)d_out, DIM, ROWS, DIM, DIM, stream);
}
